// MODEL_25426206392588
// MI455X (gfx1250) — compile-verified, matrix operand order checked by reading, not run on hardware
//
#include <hip/hip_runtime.h>
#include <hip/hip_bf16.h>

#define Bn   2
#define Sn   512
#define Vn   32000
#define Dn   1024
#define Hn   16
#define DHn  64
#define DFFn 4096
#define LEn  2
#define LDn  2
#define BSn  (Bn * Sn)

typedef __attribute__((ext_vector_type(16))) __bf16 v16bf;
typedef __attribute__((ext_vector_type(8)))  float  v8f;

__device__ __forceinline__ unsigned short f2bf(float f) {
  unsigned u = __float_as_uint(f);
  u += 0x7FFFu + ((u >> 16) & 1u);   // round-to-nearest-even bf16
  return (unsigned short)(u >> 16);
}
__device__ __forceinline__ unsigned pack2(float a, float b) {
  return (unsigned)f2bf(a) | ((unsigned)f2bf(b) << 16);
}

union Frag { uint4 q[2]; v16bf v; };

// ---------------------------------------------------------------------------
// Batched GEMM: C = act(A[M,K] * B[K,N] + bias), bf16 WMMA, f32 accumulate.
// Block: 256 threads = 8 waves; block tile 128(M) x 64(N); K-step 32.
// Wave (wr, wc) owns a 32x32 macro-tile -> 4 WMMAs/K-step.
// LDS tiles K-contiguous, 40-short row stride (80B = 5*16B) -> fragments are
// exactly two aligned ds_load_b128 per lane.
// TRANSB=1: B stored [N][K] row-major (multiply by B^T).
// ---------------------------------------------------------------------------
template <int TRANSB>
__global__ __launch_bounds__(256) void wmma_gemm_kernel(
    const float* __restrict__ A,  long aB, long aH, int lda,
    const float* __restrict__ Bm, long bB, long bH, int ldb,
    float* __restrict__ C,        long cB, long cH, int ldc,
    const float* __restrict__ bias, int K, int hdiv, int relu)
{
  const int z  = blockIdx.z;
  const int bo = z / hdiv, ho = z % hdiv;
  A  += (long)bo * aB + (long)ho * aH;
  Bm += (long)bo * bB + (long)ho * bH;
  C  += (long)bo * cB + (long)ho * cH;

  __shared__ __align__(16) unsigned short As[128][40];
  __shared__ __align__(16) unsigned short Bt[64][40];

  const int tid   = threadIdx.x;
  const int mBase = blockIdx.y * 128;
  const int nBase = blockIdx.x * 64;
  const int wave  = tid >> 5;
  const int lane  = tid & 31;
  const int wr    = wave >> 1;
  const int wc    = wave & 1;
  const int rc    = lane & 15;
  const int kb    = (lane < 16) ? 0 : 8;

  v8f acc[2][2];
  #pragma unroll
  for (int mi = 0; mi < 2; ++mi)
    #pragma unroll
    for (int ni = 0; ni < 2; ++ni)
      acc[mi][ni] = (v8f){0.f,0.f,0.f,0.f,0.f,0.f,0.f,0.f};

  for (int kt = 0; kt < K; kt += 32) {
    #pragma unroll
    for (int i = 0; i < 4; ++i) {          // A: 128x32 = 1024 float4 groups
      int idx = tid + i * 256;
      int r = idx >> 3, g = idx & 7;
      const float4 f = *(const float4*)(A + (long)(mBase + r) * lda + kt + g * 4);
      unsigned long long p = (unsigned long long)pack2(f.x, f.y)
                           | ((unsigned long long)pack2(f.z, f.w) << 32);
      *(unsigned long long*)(&As[r][g * 4]) = p;
    }
    if (TRANSB) {
      #pragma unroll
      for (int i = 0; i < 2; ++i) {        // B: 64x32 = 512 float4 groups
        int idx = tid + i * 256;
        int n = idx >> 3, g = idx & 7;
        const float4 f = *(const float4*)(Bm + (long)(nBase + n) * ldb + kt + g * 4);
        unsigned long long p = (unsigned long long)pack2(f.x, f.y)
                             | ((unsigned long long)pack2(f.z, f.w) << 32);
        *(unsigned long long*)(&Bt[n][g * 4]) = p;
      }
    } else {
      #pragma unroll
      for (int i = 0; i < 2; ++i) {        // 32 k x 16 n-groups
        int idx = tid + i * 256;
        int k = idx >> 4, g = idx & 15;
        const float4 f = *(const float4*)(Bm + (long)(kt + k) * ldb + nBase + g * 4);
        Bt[g * 4 + 0][k] = f2bf(f.x);
        Bt[g * 4 + 1][k] = f2bf(f.y);
        Bt[g * 4 + 2][k] = f2bf(f.z);
        Bt[g * 4 + 3][k] = f2bf(f.w);
      }
    }
    if (kt + 32 < K) {                     // global_prefetch next K tile
      __builtin_prefetch(A + (long)(mBase + (tid >> 1)) * lda + kt + 32, 0, 3);
      __builtin_prefetch(TRANSB ? Bm + (long)(nBase + (tid & 63)) * ldb + kt + 32
                                : Bm + (long)(kt + 32 + (tid & 31)) * ldb + nBase, 0, 3);
    }
    __syncthreads();

    Frag af[2], bf[2];
    #pragma unroll
    for (int mi = 0; mi < 2; ++mi) {
      int row = wr * 32 + mi * 16 + rc;
      af[mi].q[0] = *(const uint4*)(&As[row][kb]);
      af[mi].q[1] = *(const uint4*)(&As[row][kb + 16]);
    }
    #pragma unroll
    for (int ni = 0; ni < 2; ++ni) {
      int nrow = wc * 32 + ni * 16 + rc;
      bf[ni].q[0] = *(const uint4*)(&Bt[nrow][kb]);
      bf[ni].q[1] = *(const uint4*)(&Bt[nrow][kb + 16]);
    }
    #pragma unroll
    for (int mi = 0; mi < 2; ++mi)
      #pragma unroll
      for (int ni = 0; ni < 2; ++ni)
        acc[mi][ni] = __builtin_amdgcn_wmma_f32_16x16x32_bf16(
            false, af[mi].v, false, bf[ni].v, (short)0, acc[mi][ni], false, false);
    __syncthreads();
  }

  #pragma unroll
  for (int mi = 0; mi < 2; ++mi)
    #pragma unroll
    for (int ni = 0; ni < 2; ++ni)
      #pragma unroll
      for (int r8 = 0; r8 < 8; ++r8) {
        int m = mBase + wr * 32 + mi * 16 + r8 + ((lane < 16) ? 0 : 8);
        int n = nBase + wc * 32 + ni * 16 + rc;
        float v = acc[mi][ni][r8];
        if (bias) v += bias[n];
        if (relu) v = fmaxf(v, 0.f);
        C[(long)m * ldc + n] = v;
      }
}

// ---------------------------------------------------------------------------
// GEMM variant for attn_probs(bf16) x V(f32): A is already bf16 in memory, so
// the A tile is staged by the CDNA5 async DMA path (global_load_async_to_lds)
// with s_wait_asynccnt, no conversion VALU. Same 128x64 tile, 4 WMMAs/K-step.
// ---------------------------------------------------------------------------
__global__ __launch_bounds__(256) void wmma_gemm_bf16A_kernel(
    const unsigned short* __restrict__ Ah, long aB, long aH, int lda,
    const float* __restrict__ Bm, long bB, long bH, int ldb,
    float* __restrict__ C, long cB, long cH, int ldc,
    int K, int hdiv)
{
  const int z  = blockIdx.z;
  const int bo = z / hdiv, ho = z % hdiv;
  Ah += (long)bo * aB + (long)ho * aH;
  Bm += (long)bo * bB + (long)ho * bH;
  C  += (long)bo * cB + (long)ho * cH;

  __shared__ __align__(16) unsigned short As[128][40];
  __shared__ __align__(16) unsigned short Bt[64][40];

  const int tid   = threadIdx.x;
  const int mBase = blockIdx.y * 128;
  const int nBase = blockIdx.x * 64;
  const int wave  = tid >> 5;
  const int lane  = tid & 31;
  const int wr    = wave >> 1;
  const int wc    = wave & 1;
  const int rc    = lane & 15;
  const int kb    = (lane < 16) ? 0 : 8;

  v8f acc[2][2];
  #pragma unroll
  for (int mi = 0; mi < 2; ++mi)
    #pragma unroll
    for (int ni = 0; ni < 2; ++ni)
      acc[mi][ni] = (v8f){0.f,0.f,0.f,0.f,0.f,0.f,0.f,0.f};

  for (int kt = 0; kt < K; kt += 32) {
    // ---- A tile: async DMA global(bf16) -> LDS, 16B per issue, 2/thread ----
    #pragma unroll
    for (int i = 0; i < 2; ++i) {
      int idx = tid + i * 256;
      int r = idx >> 2, c = idx & 3;                     // 128 rows x 4 chunks
      unsigned lds = (unsigned)(unsigned long long)(void*)&As[r][c * 8];
      unsigned long long ga =
          (unsigned long long)(Ah + (long)(mBase + r) * lda + kt + c * 8);
      asm volatile("global_load_async_to_lds_b128 %0, %1, off"
                   :: "v"(lds), "v"(ga) : "memory");
    }
    // ---- B tile: f32 -> bf16, n-major ----
    #pragma unroll
    for (int i = 0; i < 2; ++i) {
      int idx = tid + i * 256;
      int k = idx >> 4, g = idx & 15;
      const float4 f = *(const float4*)(Bm + (long)(kt + k) * ldb + nBase + g * 4);
      Bt[g * 4 + 0][k] = f2bf(f.x);
      Bt[g * 4 + 1][k] = f2bf(f.y);
      Bt[g * 4 + 2][k] = f2bf(f.z);
      Bt[g * 4 + 3][k] = f2bf(f.w);
    }
    asm volatile("s_wait_asynccnt 0" ::: "memory");
    __syncthreads();

    Frag af[2], bf[2];
    #pragma unroll
    for (int mi = 0; mi < 2; ++mi) {
      int row = wr * 32 + mi * 16 + rc;
      af[mi].q[0] = *(const uint4*)(&As[row][kb]);
      af[mi].q[1] = *(const uint4*)(&As[row][kb + 16]);
    }
    #pragma unroll
    for (int ni = 0; ni < 2; ++ni) {
      int nrow = wc * 32 + ni * 16 + rc;
      bf[ni].q[0] = *(const uint4*)(&Bt[nrow][kb]);
      bf[ni].q[1] = *(const uint4*)(&Bt[nrow][kb + 16]);
    }
    #pragma unroll
    for (int mi = 0; mi < 2; ++mi)
      #pragma unroll
      for (int ni = 0; ni < 2; ++ni)
        acc[mi][ni] = __builtin_amdgcn_wmma_f32_16x16x32_bf16(
            false, af[mi].v, false, bf[ni].v, (short)0, acc[mi][ni], false, false);
    __syncthreads();
  }

  #pragma unroll
  for (int mi = 0; mi < 2; ++mi)
    #pragma unroll
    for (int ni = 0; ni < 2; ++ni)
      #pragma unroll
      for (int r8 = 0; r8 < 8; ++r8) {
        int m = mBase + wr * 32 + mi * 16 + r8 + ((lane < 16) ? 0 : 8);
        int n = nBase + wc * 32 + ni * 16 + rc;
        C[(long)m * ldc + n] = acc[mi][ni][r8];
      }
}

// ---------------------------------------------------------------------------
__global__ void embed_kernel(const int* __restrict__ ids,
                             const float* __restrict__ tab,
                             float* __restrict__ out)
{
  int t = blockIdx.x;
  const float* src = tab + (long)ids[t] * Dn;
  float* dst = out + (long)t * Dn;
  for (int c = threadIdx.x; c < Dn; c += blockDim.x) dst[c] = src[c];
}

// scale 1/sqrt(64), mask (mode0: pad[k]==0 ; mode1: pad[k]==0 && k<=q),
// softmax over k; emit normalized probabilities as packed bf16 into Pb.
__global__ __launch_bounds__(256) void score_softmax_kernel(
    const float* __restrict__ SC, unsigned short* __restrict__ Pb,
    const int* __restrict__ pad, int mode)
{
  int r  = blockIdx.x;                  // (b*H + h)*S + q
  int q  = r % Sn;
  int b  = (r / Sn) / Hn;
  const float* row = SC + (long)r * Sn;
  unsigned* prow = (unsigned*)(Pb + (long)r * Sn);
  int tid = threadIdx.x;
  __shared__ float red[256];

  float v[2];
  #pragma unroll
  for (int i = 0; i < 2; ++i) {
    int k = 2 * tid + i;
    float s = row[k] * 0.125f;
    bool masked = (pad[b * Sn + k] == 0) && (mode == 0 || k <= q);
    v[i] = masked ? -1e9f : s;
  }
  float m = fmaxf(v[0], v[1]);
  red[tid] = m; __syncthreads();
  for (int o = 128; o > 0; o >>= 1) { if (tid < o) red[tid] = fmaxf(red[tid], red[tid + o]); __syncthreads(); }
  m = red[0]; __syncthreads();
  float e0 = __expf(v[0] - m), e1 = __expf(v[1] - m);
  red[tid] = e0 + e1; __syncthreads();
  for (int o = 128; o > 0; o >>= 1) { if (tid < o) red[tid] += red[tid + o]; __syncthreads(); }
  float inv = 1.f / red[0];
  prow[tid] = pack2(e0 * inv, e1 * inv);
}

// out = LN(x + h) * g + b  (row length D=1024; out may alias x)
__global__ __launch_bounds__(256) void add_ln_kernel(
    const float* __restrict__ x, const float* __restrict__ h,
    const float* __restrict__ g, const float* __restrict__ bta,
    float* __restrict__ out)
{
  int t = blockIdx.x, tid = threadIdx.x;
  const float* xr = x + (long)t * Dn;
  const float* hr = h + (long)t * Dn;
  float* orow = out + (long)t * Dn;
  __shared__ float red[256];
  float v[4], s = 0.f;
  #pragma unroll
  for (int i = 0; i < 4; ++i) { int c = tid + i * 256; v[i] = xr[c] + hr[c]; s += v[i]; }
  red[tid] = s; __syncthreads();
  for (int o = 128; o > 0; o >>= 1) { if (tid < o) red[tid] += red[tid + o]; __syncthreads(); }
  float mu = red[0] * (1.f / Dn); __syncthreads();
  s = 0.f;
  #pragma unroll
  for (int i = 0; i < 4; ++i) { float d = v[i] - mu; s += d * d; }
  red[tid] = s; __syncthreads();
  for (int o = 128; o > 0; o >>= 1) { if (tid < o) red[tid] += red[tid + o]; __syncthreads(); }
  float rstd = rsqrtf(red[0] * (1.f / Dn) + 1e-5f);
  #pragma unroll
  for (int i = 0; i < 4; ++i) {
    int c = tid + i * 256;
    orow[c] = (v[i] - mu) * rstd * g[c] + bta[c];
  }
}

__global__ void zero_loss_kernel(float* out) {
  if (threadIdx.x == 0 && blockIdx.x == 0) out[0] = 0.f;
}

// loss = -(1/(B*V)) sum_{b,s} (logits[b,s,lab] - logsumexp_{s'} logits[b,s',lab])
__global__ __launch_bounds__(256) void loss_kernel(
    const float* __restrict__ logits, const int* __restrict__ labels,
    float* __restrict__ loss)
{
  int t = blockIdx.x, tid = threadIdx.x;
  int b = t / Sn, s = t % Sn;
  int v = labels[t];
  const float* col = logits + (long)b * Sn * Vn + v;   // stride Vn over s'
  __shared__ float red[256];
  float a0 = col[(long)tid * Vn];
  float a1 = col[(long)(tid + 256) * Vn];
  float m = fmaxf(a0, a1);
  red[tid] = m; __syncthreads();
  for (int o = 128; o > 0; o >>= 1) { if (tid < o) red[tid] = fmaxf(red[tid], red[tid + o]); __syncthreads(); }
  m = red[0]; __syncthreads();
  red[tid] = __expf(a0 - m) + __expf(a1 - m); __syncthreads();
  for (int o = 128; o > 0; o >>= 1) { if (tid < o) red[tid] += red[tid + o]; __syncthreads(); }
  if (tid == 0) {
    float lse = m + __logf(red[0]);
    float lv  = logits[((long)b * Sn + s) * Vn + v];
    atomicAdd(loss, -(lv - lse) * (1.f / (Bn * (float)Vn)));
  }
}

// softmax over V=32000 in place (probs output)
__global__ __launch_bounds__(256) void softmaxV_kernel(float* __restrict__ p)
{
  int t = blockIdx.x, tid = threadIdx.x;
  float* row = p + (long)t * Vn;
  __shared__ float red[256];
  float m = -1e30f;
  for (int c = tid; c < Vn; c += 256) m = fmaxf(m, row[c]);
  red[tid] = m; __syncthreads();
  for (int o = 128; o > 0; o >>= 1) { if (tid < o) red[tid] = fmaxf(red[tid], red[tid + o]); __syncthreads(); }
  m = red[0]; __syncthreads();
  float s = 0.f;
  for (int c = tid; c < Vn; c += 256) s += __expf(row[c] - m);
  red[tid] = s; __syncthreads();
  for (int o = 128; o > 0; o >>= 1) { if (tid < o) red[tid] += red[tid + o]; __syncthreads(); }
  float inv = 1.f / red[0];
  for (int c = tid; c < Vn; c += 256) row[c] = __expf(row[c] - m) * inv;
}

// ---------------------------------------------------------------------------
static void launch_gemm(hipStream_t st,
    const float* A, long aB, long aH, int lda,
    const float* Bm, long bB, long bH, int ldb, int transB,
    float* C, long cB, long cH, int ldc,
    const float* bias, int M, int N, int K, int batch, int hdiv, int relu)
{
  dim3 g(N / 64, M / 128, batch);
  if (transB)
    wmma_gemm_kernel<1><<<g, 256, 0, st>>>(A, aB, aH, lda, Bm, bB, bH, ldb,
                                           C, cB, cH, ldc, bias, K, hdiv, relu);
  else
    wmma_gemm_kernel<0><<<g, 256, 0, st>>>(A, aB, aH, lda, Bm, bB, bH, ldb,
                                           C, cB, cH, ldc, bias, K, hdiv, relu);
}

static void run_attention(hipStream_t st,
    const float* qin, const float* kvin,
    const float* wq, const float* bq, const float* wk, const float* bk,
    const float* wv, const float* bv, const float* wo, const float* bo,
    const int* padmask, int maskmode,
    float* Q, float* Kb, float* Vb, float* O, float* T, float* SC,
    unsigned short* Pb)
{
  launch_gemm(st, qin, 0, 0, Dn, wq, 0, 0, Dn, 0, Q,  0, 0, Dn, bq, BSn, Dn, Dn, 1, 1, 0);
  launch_gemm(st, kvin,0, 0, Dn, wk, 0, 0, Dn, 0, Kb, 0, 0, Dn, bk, BSn, Dn, Dn, 1, 1, 0);
  launch_gemm(st, kvin,0, 0, Dn, wv, 0, 0, Dn, 0, Vb, 0, 0, Dn, bv, BSn, Dn, Dn, 1, 1, 0);
  // scores[b,h] = Q_bh (S x 64) * K_bh^T (64 x S)
  launch_gemm(st, Q,  (long)Sn * Dn, DHn, Dn,
                  Kb, (long)Sn * Dn, DHn, Dn, 1,
                  SC, (long)Hn * Sn * Sn, (long)Sn * Sn, Sn,
                  nullptr, Sn, Sn, DHn, Bn * Hn, Hn, 0);
  score_softmax_kernel<<<Bn * Hn * Sn, 256, 0, st>>>(SC, Pb, padmask, maskmode);
  // O[b,h] = probs_bf16 (S x S) * V_bh (S x 64)  -- async-DMA A staging
  {
    dim3 g(DHn / 64, Sn / 128, Bn * Hn);
    wmma_gemm_bf16A_kernel<<<g, 256, 0, st>>>(
        Pb, (long)Hn * Sn * Sn, (long)Sn * Sn, Sn,
        Vb, (long)Sn * Dn, DHn, Dn,
        O,  (long)Sn * Dn, DHn, Dn, Sn, Hn);
  }
  launch_gemm(st, O, 0, 0, Dn, wo, 0, 0, Dn, 0, T, 0, 0, Dn, bo, BSn, Dn, Dn, 1, 1, 0);
}

static void run_mlp(hipStream_t st, const float* xin,
    const float* w1, const float* b1, const float* w2, const float* b2,
    float* H1, float* T)
{
  launch_gemm(st, xin, 0, 0, Dn,   w1, 0, 0, DFFn, 0, H1, 0, 0, DFFn, b1, BSn, DFFn, Dn,   1, 1, 1);
  launch_gemm(st, H1,  0, 0, DFFn, w2, 0, 0, Dn,   0, T,  0, 0, Dn,   b2, BSn, Dn,   DFFn, 1, 1, 0);
}

extern "C" void kernel_launch(void* const* d_in, const int* in_sizes, int n_in,
                              void* d_out, int out_size, void* d_ws, size_t ws_size,
                              hipStream_t stream)
{
  (void)in_sizes; (void)n_in; (void)out_size; (void)ws_size;
  const int* input_ids = (const int*)d_in[0];
  const int* label_ids = (const int*)d_in[1];
  const int* in_pad    = (const int*)d_in[2];
  const int* tg_pad    = (const int*)d_in[3];
  const float* emb_in  = (const float*)d_in[4];
  const float* emb_tgt = (const float*)d_in[5];
  auto F = [&](int i) { return (const float*)d_in[i]; };
  // d_in layout (dict order): 6..13 enc swq,sbq,swk,sbk,swv,sbv,swo,sbo
  // 14..17 enc w1,b1,w2,b2 ; 18..21 enc ln1s,ln1b,ln2s,ln2b
  // 22..29 dec self qkvo ; 30..37 dec cross qkvo ; 38..41 dec w1,b1,w2,b2
  // 42..47 dec ln1s..ln3b ; 48 cls_w ; 49 cls_b

  float* ws = (float*)d_ws;
  const long M1 = (long)BSn * Dn;          // 1,048,576 floats
  float* X  = ws;
  float* Y  = ws + 1 * M1;
  float* Q  = ws + 2 * M1;
  float* Kb = ws + 3 * M1;
  float* Vb = ws + 4 * M1;
  float* O  = ws + 5 * M1;
  float* T  = ws + 6 * M1;
  float* H1 = ws + 7 * M1;                 // BSn*DFFn = 4*M1 floats
  float* SC = ws + 11 * M1;                // B*H*S*S  = 8*M1 floats
  unsigned short* Pb = (unsigned short*)(ws + 19 * M1);  // bf16 probs, 4*M1 floats

  float* outF   = (float*)d_out;
  float* logits = outF + 1;                // probs region, logits first

  embed_kernel<<<BSn, 256, 0, stream>>>(input_ids, emb_in,  X);
  embed_kernel<<<BSn, 256, 0, stream>>>(label_ids, emb_tgt, Y);

  for (int l = 0; l < LEn; ++l) {
    long wO = (long)l * Dn * Dn, bO = (long)l * Dn;
    run_attention(stream, X, X,
        F(6) + wO, F(7) + bO, F(8) + wO, F(9) + bO,
        F(10) + wO, F(11) + bO, F(12) + wO, F(13) + bO,
        in_pad, 0, Q, Kb, Vb, O, T, SC, Pb);
    add_ln_kernel<<<BSn, 256, 0, stream>>>(X, T, F(18) + bO, F(19) + bO, X);
    run_mlp(stream, X, F(14) + (long)l * Dn * DFFn, F(15) + (long)l * DFFn,
            F(16) + (long)l * DFFn * Dn, F(17) + bO, H1, T);
    add_ln_kernel<<<BSn, 256, 0, stream>>>(X, T, F(20) + bO, F(21) + bO, X);
  }

  for (int l = 0; l < LDn; ++l) {
    long wO = (long)l * Dn * Dn, bO = (long)l * Dn;
    run_attention(stream, Y, Y,
        F(22) + wO, F(23) + bO, F(24) + wO, F(25) + bO,
        F(26) + wO, F(27) + bO, F(28) + wO, F(29) + bO,
        tg_pad, 1, Q, Kb, Vb, O, T, SC, Pb);
    add_ln_kernel<<<BSn, 256, 0, stream>>>(Y, T, F(42) + bO, F(43) + bO, Y);
    run_attention(stream, Y, X,
        F(30) + wO, F(31) + bO, F(32) + wO, F(33) + bO,
        F(34) + wO, F(35) + bO, F(36) + wO, F(37) + bO,
        in_pad, 0, Q, Kb, Vb, O, T, SC, Pb);
    add_ln_kernel<<<BSn, 256, 0, stream>>>(Y, T, F(44) + bO, F(45) + bO, Y);
    run_mlp(stream, Y, F(38) + (long)l * Dn * DFFn, F(39) + (long)l * DFFn,
            F(40) + (long)l * DFFn * Dn, F(41) + bO, H1, T);
    add_ln_kernel<<<BSn, 256, 0, stream>>>(Y, T, F(46) + bO, F(47) + bO, Y);
  }

  // logits = Y @ cls_w + cls_b   (1024 x 32000, K=1024)
  launch_gemm(stream, Y, 0, 0, Dn, F(48), 0, 0, Vn, 0,
              logits, 0, 0, Vn, F(49), BSn, Vn, Dn, 1, 1, 0);
  zero_loss_kernel<<<1, 32, 0, stream>>>(outF);
  loss_kernel<<<BSn, 256, 0, stream>>>(logits, label_ids, outF);
  softmaxV_kernel<<<BSn, 256, 0, stream>>>(logits);   // in-place -> probs
}
